// GeneratorNCA_77360950936230
// MI455X (gfx1250) — matrix operand order repaired and checked by reading, not compiled, not run on hardware
//
#include <hip/hip_runtime.h>
#include <hip/hip_bf16.h>
#include <math.h>

// MI455X / gfx1250, wave32. Matrix path: v_wmma_f32_16x16x32_f16.
// Async path: global_load_async_to_lds_b128 + s_wait_asynccnt (ASYNCcnt).
typedef __attribute__((ext_vector_type(16))) _Float16 v16h;
typedef __attribute__((ext_vector_type(8)))  _Float16 v8h;
typedef __attribute__((ext_vector_type(8)))  float    v8f;

#define B_     8
#define C_     16
#define L_     32
#define HW_    16
#define T_     128
#define HID_   128
#define STEPS_ 8

__device__ __forceinline__ unsigned pcg_hash(unsigned v) {
    unsigned s = v * 747796405u + 2891336453u;
    unsigned w = ((s >> ((s >> 28) + 4u)) ^ s) * 277803737u;
    return (w >> 22) ^ w;
}

// ---------------------------------------------------------------------------
// Bilinear upsample z_field (B,L,16,16) f32 -> z_up (B,L,128,128) f16
// ---------------------------------------------------------------------------
__global__ __launch_bounds__(256) void nca_upsample(const float* __restrict__ z,
                                                    _Float16* __restrict__ zupH) {
    int idx = blockIdx.x * 256 + threadIdx.x;          // 4,194,304
    int x = idx & 127;
    int y = (idx >> 7) & 127;
    int l = (idx >> 14) & 31;
    int b = idx >> 19;
    float fy = (y + 0.5f) * 0.125f - 0.5f;
    float fx = (x + 0.5f) * 0.125f - 0.5f;
    int y0 = (int)floorf(fy); float ty = fy - (float)y0;
    int x0 = (int)floorf(fx); float tx = fx - (float)x0;
    int y0c = min(max(y0, 0), HW_ - 1), y1c = min(max(y0 + 1, 0), HW_ - 1);
    int x0c = min(max(x0, 0), HW_ - 1), x1c = min(max(x0 + 1, 0), HW_ - 1);
    const float* src = z + ((size_t)(b * L_ + l)) * HW_ * HW_;
    float v00 = src[y0c * HW_ + x0c], v01 = src[y0c * HW_ + x1c];
    float v10 = src[y1c * HW_ + x0c], v11 = src[y1c * HW_ + x1c];
    float v = (1.f - ty) * ((1.f - tx) * v00 + tx * v01) +
              ty         * ((1.f - tx) * v10 + tx * v11);
    zupH[idx] = (_Float16)v;
}

// ---------------------------------------------------------------------------
// Pre-swizzle weights f32 -> f16 into WMMA B-fragment order:
// fragment(nt,kt): 32 lanes x 16 contiguous halves, element j of lane ->
//   B[k = kt*32 + 8*(lane>>4) + (j&7) + 16*(j>>3)][n = n0 + (lane&15)]
// so each lane loads its fragment as two global_load_b128.
// ---------------------------------------------------------------------------
__global__ __launch_bounds__(256) void nca_wconv(const float* __restrict__ w1,
                                                 const float* __restrict__ w2,
                                                 _Float16* __restrict__ w1s,
                                                 _Float16* __restrict__ w2s) {
    int i = blockIdx.x * 256 + threadIdx.x;            // 14336 total
    if (i < 96 * HID_) {                               // 24 fragments * 512
        int j    = i & 15;
        int lane = (i >> 4) & 31;
        int f    = i >> 9;                             // 0..23
        int nt   = f / 3;
        int kt   = f - 3 * nt;
        int k    = kt * 32 + ((lane >> 4) << 3) + ((j & 8) << 1) + (j & 7);
        w1s[i] = (_Float16)w1[k * HID_ + nt * 16 + (lane & 15)];
    } else if (i < 96 * HID_ + HID_ * C_) {            // 4 fragments * 512
        int i2   = i - 96 * HID_;
        int j    = i2 & 15;
        int lane = (i2 >> 4) & 31;
        int kt   = i2 >> 9;                            // 0..3
        int k    = kt * 32 + ((lane >> 4) << 3) + ((j & 8) << 1) + (j & 7);
        w2s[i2] = (_Float16)w2[k * C_ + (lane & 15)];
    }
}

// ---------------------------------------------------------------------------
// Init state: zeros, channels 3..15 at (64,64) = 1.0   (float4 stores)
// ---------------------------------------------------------------------------
__global__ __launch_bounds__(256) void nca_init(float* __restrict__ stateA) {
    int i4 = blockIdx.x * 256 + threadIdx.x;           // 524288
    int base = i4 * 4;
    int yx = base & (T_ * T_ - 1);
    int c = (base >> 14) & 15;
    float4 v = make_float4(0.f, 0.f, 0.f, 0.f);
    if (c >= 3 && yx == (T_ / 2) * T_ + (T_ / 2)) v.x = 1.f;  // 8256 % 4 == 0
    *(float4*)&stateA[base] = v;
}

// ---------------------------------------------------------------------------
// Fused NCA step. One block per (b, y) row; 8 waves x 16-pixel M-tiles.
// ---------------------------------------------------------------------------
__global__ __launch_bounds__(256) void nca_step(const float* __restrict__ stateA,
                                                float* __restrict__ stateB,
                                                const _Float16* __restrict__ zupH,
                                                const _Float16* __restrict__ w1s,
                                                const _Float16* __restrict__ w2s,
                                                const float* __restrict__ b1,
                                                const float* __restrict__ b2,
                                                int step) {
    // X[128][64] f16: state + perception features (k = 0..63), row-major.
    // Pool: first sst[3][16][130] staged state rows, then per-wave H[16][128].
    // zrow[32][128] f16: this row's z_up slice, filled by async DMA (k-major).
    __shared__ __align__(16) _Float16 Xl[T_ * 64];          // 16384 B
    __shared__ __align__(16) _Float16 Pool[8 * 16 * HID_];  // 32768 B
    __shared__ __align__(16) _Float16 zrow[L_ * T_];        //  8192 B

    const int tid = threadIdx.x;
    const int y = blockIdx.x;
    const int b = blockIdx.y;

    __builtin_prefetch(w1s, 0, 1);   // global_prefetch_b8: weights are L2-hot

    // --- Kick off async DMA of z_up row slice into LDS (ASYNCcnt path).
    // 512 x 16B chunks; zrow is k-major so global reads are fully coalesced.
    // Low 32 bits of a flat LDS pointer are the wave-relative LDS address
    // (ISA 10.2: LDS aperture).
#pragma unroll
    for (int it = 0; it < 2; ++it) {
        int q = tid + it * 256;                         // chunk id, 0..511
        int l = q >> 4;
        unsigned lds_addr = (unsigned)(size_t)&zrow[q * 8];
        unsigned long long gaddr =
            (unsigned long long)(size_t)&zupH[((b * L_ + l) * T_ + y) * T_ + (q & 15) * 8];
        asm volatile("global_load_async_to_lds_b128 %0, %1, off"
                     :: "v"(lds_addr), "v"(gaddr) : "memory");
    }

    // --- Phase A: stage 3 state rows (y-1,y,y+1) x 16 ch, zero halo, f16
    for (int e = tid; e < 3 * 16 * 130; e += 256) {
        int r   = e / (16 * 130);
        int rem = e - r * (16 * 130);
        int c   = rem / 130;
        int xi  = rem - c * 130;
        int gx = xi - 1;
        int gy = y + r - 1;
        float v = 0.f;
        if (gx >= 0 && gx < T_ && gy >= 0 && gy < T_)
            v = stateA[((b * C_ + c) * T_ + gy) * T_ + gx];
        Pool[e] = (_Float16)v;
    }
    __syncthreads();

    // --- Phase B: features X[px][0:16]=state, [16:64]=Sobel-x/y + Laplacian
    for (int e = tid; e < T_ * C_; e += 256) {
        int c  = e & 15;
        int px = e >> 4;
        float s00 = (float)Pool[(0 * 16 + c) * 130 + px + 0];
        float s01 = (float)Pool[(0 * 16 + c) * 130 + px + 1];
        float s02 = (float)Pool[(0 * 16 + c) * 130 + px + 2];
        float s10 = (float)Pool[(1 * 16 + c) * 130 + px + 0];
        float s11 = (float)Pool[(1 * 16 + c) * 130 + px + 1];
        float s12 = (float)Pool[(1 * 16 + c) * 130 + px + 2];
        float s20 = (float)Pool[(2 * 16 + c) * 130 + px + 0];
        float s21 = (float)Pool[(2 * 16 + c) * 130 + px + 1];
        float s22 = (float)Pool[(2 * 16 + c) * 130 + px + 2];
        float sx  = (s02 - s00) + 2.f * (s12 - s10) + (s22 - s20);
        float sy  = (s20 + 2.f * s21 + s22) - (s00 + 2.f * s01 + s02);
        float lap = (s00 + 2.f * s01 + s02 + 2.f * s10 - 12.f * s11 + 2.f * s12 +
                     s20 + 2.f * s21 + s22) * 0.125f;
        _Float16* xp = &Xl[px * 64];
        xp[c]              = (_Float16)s11;
        xp[16 + 3 * c + 0] = (_Float16)sx;
        xp[16 + 3 * c + 1] = (_Float16)sy;
        xp[16 + 3 * c + 2] = (_Float16)lap;
    }

    const int wave = tid >> 5;
    const int lane = tid & 31;
    const int m  = lane & 15;          // A-fragment row
    const int hi = lane >> 4;          // lane-group select
    const int c  = lane & 15;          // D-fragment column
    const int waveBase = wave * 16;

    // preload old state (center row) before Pool is reused as H
    float sold[8];
#pragma unroll
    for (int r = 0; r < 8; ++r)
        sold[r] = (float)Pool[(1 * 16 + c) * 130 + waveBase + r + 8 * hi + 1];

    asm volatile("s_wait_asynccnt 0" ::: "memory");   // zrow DMA complete
    __syncthreads();                                  // X/sold done; Pool -> H

    // --- Phase C: layer 1.  A-fragments are nt-invariant: hoist them.
    // ISA 16-bit A 16x32 layout per lane = two contiguous 8-half runs.
    const _Float16* Xw = &Xl[waveBase * 64];
    v16h a0, a1, a2;
    {
        v8h lo, hh;
        lo = *(const v8h*)&Xw[m * 64 + 0  + (hi << 3)];
        hh = *(const v8h*)&Xw[m * 64 + 16 + (hi << 3)];
        a0 = __builtin_shufflevector(lo, hh, 0,1,2,3,4,5,6,7,8,9,10,11,12,13,14,15);
        lo = *(const v8h*)&Xw[m * 64 + 32 + (hi << 3)];
        hh = *(const v8h*)&Xw[m * 64 + 48 + (hi << 3)];
        a1 = __builtin_shufflevector(lo, hh, 0,1,2,3,4,5,6,7,8,9,10,11,12,13,14,15);
        // kt=2 (z_up): gather from k-major zrow once per wave
        const int pxm = waveBase + m;
#pragma unroll
        for (int i = 0; i < 16; ++i) {
            int l = (((i & 8) << 1) | (i & 7)) + (hi << 3);
            a2[i] = zrow[l * T_ + pxm];
        }
    }

    _Float16* Hw = &Pool[wave * (16 * HID_)];
#pragma unroll
    for (int nt = 0; nt < HID_ / 16; ++nt) {
        const _Float16* wp = &w1s[(nt * 3) << 9];
        v8h lo, hh;
        v16h bf;
        v8f acc = {};
        lo = *(const v8h*)&wp[(0 << 9) + (lane << 4)];
        hh = *(const v8h*)&wp[(0 << 9) + (lane << 4) + 8];
        bf = __builtin_shufflevector(lo, hh, 0,1,2,3,4,5,6,7,8,9,10,11,12,13,14,15);
        acc = __builtin_amdgcn_wmma_f32_16x16x32_f16(false, a0, false, bf, (short)0, acc, false, false);
        lo = *(const v8h*)&wp[(1 << 9) + (lane << 4)];
        hh = *(const v8h*)&wp[(1 << 9) + (lane << 4) + 8];
        bf = __builtin_shufflevector(lo, hh, 0,1,2,3,4,5,6,7,8,9,10,11,12,13,14,15);
        acc = __builtin_amdgcn_wmma_f32_16x16x32_f16(false, a1, false, bf, (short)0, acc, false, false);
        lo = *(const v8h*)&wp[(2 << 9) + (lane << 4)];
        hh = *(const v8h*)&wp[(2 << 9) + (lane << 4) + 8];
        bf = __builtin_shufflevector(lo, hh, 0,1,2,3,4,5,6,7,8,9,10,11,12,13,14,15);
        acc = __builtin_amdgcn_wmma_f32_16x16x32_f16(false, a2, false, bf, (short)0, acc, false, false);

        float bias = b1[nt * 16 + c];
#pragma unroll
        for (int r = 0; r < 8; ++r) {
            float v = acc[r] + bias;
            v = v > 0.f ? v : 0.f;
            Hw[(r + 8 * hi) * HID_ + nt * 16 + c] = (_Float16)v;   // D: M=r+8*hi, N=c
        }
    }

    // --- Phase D: layer 2 (4 WMMAs), H row-major -> A fragments are 2xb128
    v8f acc2 = {};
#pragma unroll
    for (int kt = 0; kt < HID_ / 32; ++kt) {
        v8h lo, hh;
        lo = *(const v8h*)&Hw[m * HID_ + kt * 32 + 0  + (hi << 3)];
        hh = *(const v8h*)&Hw[m * HID_ + kt * 32 + 16 + (hi << 3)];
        v16h a = __builtin_shufflevector(lo, hh, 0,1,2,3,4,5,6,7,8,9,10,11,12,13,14,15);
        lo = *(const v8h*)&w2s[(kt << 9) + (lane << 4)];
        hh = *(const v8h*)&w2s[(kt << 9) + (lane << 4) + 8];
        v16h bf = __builtin_shufflevector(lo, hh, 0,1,2,3,4,5,6,7,8,9,10,11,12,13,14,15);
        acc2 = __builtin_amdgcn_wmma_f32_16x16x32_f16(false, a, false, bf, (short)0, acc2, false, false);
    }

    // --- Update: one lane owns 8 consecutive pixels of channel c -> 2x16B stores
    float bias2 = b2[c];
    float outv[8];
#pragma unroll
    for (int r = 0; r < 8; ++r) {
        int px = waveBase + r + 8 * hi;
        unsigned seed = (((unsigned)(step * B_ + b) * T_ + (unsigned)y) * T_ + (unsigned)px);
        unsigned hsh = pcg_hash(seed * 2654435761u + 0x9E3779B9u);
        float maskv = (hsh & 0x80000000u) ? 1.f : 0.f;
        outv[r] = sold[r] + (acc2[r] + bias2) * maskv;
    }
    float* dst = &stateB[((b * C_ + c) * T_ + y) * T_ + waveBase + 8 * hi];
    *(float4*)(dst + 0) = make_float4(outv[0], outv[1], outv[2], outv[3]);
    *(float4*)(dst + 4) = make_float4(outv[4], outv[5], outv[6], outv[7]);
}

// ---------------------------------------------------------------------------
// Alive mask: 3x3 max-pool of ch 3 (SAME, -inf pad); gate all channels.
// Each thread: 4 consecutive pixels, float4 loads/stores over 16 channels.
// ---------------------------------------------------------------------------
__global__ __launch_bounds__(256) void nca_alive(const float* __restrict__ stateB,
                                                 float* __restrict__ stateA) {
    int p4 = blockIdx.x * 256 + threadIdx.x;           // 32768 groups of 4 px
    int b  = p4 >> 12;
    int rem = p4 & 4095;
    int y  = rem >> 5;
    int xg = (rem & 31) * 4;
    const float* ch3 = &stateB[(b * C_ + 3) * T_ * T_];
    float keep[4];
#pragma unroll
    for (int j = 0; j < 4; ++j) {
        int x = xg + j;
        float alive = -INFINITY;
#pragma unroll
        for (int dy = -1; dy <= 1; ++dy) {
            int ny = y + dy;
            if (ny < 0 || ny >= T_) continue;
#pragma unroll
            for (int dx = -1; dx <= 1; ++dx) {
                int nx = x + dx;
                if (nx < 0 || nx >= T_) continue;
                alive = fmaxf(alive, ch3[ny * T_ + nx]);
            }
        }
        keep[j] = alive > 0.1f ? 1.f : 0.f;
    }
#pragma unroll
    for (int cc = 0; cc < C_; ++cc) {
        int idx = ((b * C_ + cc) * T_ + y) * T_ + xg;
        float4 v = *(const float4*)&stateB[idx];
        v.x *= keep[0]; v.y *= keep[1]; v.z *= keep[2]; v.w *= keep[3];
        *(float4*)&stateA[idx] = v;
    }
}

// ---------------------------------------------------------------------------
// Output: sigmoid of first 3 channels -> (B,3,128,128) f32, float4 I/O
// ---------------------------------------------------------------------------
__global__ __launch_bounds__(256) void nca_out(const float* __restrict__ stateA,
                                               float* __restrict__ out) {
    int i4 = blockIdx.x * 256 + threadIdx.x;           // 98304
    int base = i4 * 4;
    int b = base / (3 * T_ * T_);
    int rem = base - b * (3 * T_ * T_);
    int cyx = rem;                                     // c*16384 + yx
    float4 v = *(const float4*)&stateA[b * C_ * T_ * T_ + cyx];
    float4 o;
    o.x = 1.f / (1.f + __expf(-v.x));
    o.y = 1.f / (1.f + __expf(-v.y));
    o.z = 1.f / (1.f + __expf(-v.z));
    o.w = 1.f / (1.f + __expf(-v.w));
    *(float4*)&out[base] = o;
}

// ---------------------------------------------------------------------------
extern "C" void kernel_launch(void* const* d_in, const int* in_sizes, int n_in,
                              void* d_out, int out_size, void* d_ws, size_t ws_size,
                              hipStream_t stream) {
    const float* z_field = (const float*)d_in[0];   // (8,32,16,16)
    const float* w1      = (const float*)d_in[1];   // (96,128)
    const float* b1      = (const float*)d_in[2];   // (128)
    const float* w2      = (const float*)d_in[3];   // (128,16)
    const float* b2      = (const float*)d_in[4];   // (16)
    float* out           = (float*)d_out;           // (8,3,128,128)

    // workspace (~25.2 MB): stateA | stateB | zupH(f16) | w1s | w2s
    char* ws = (char*)d_ws;
    const size_t stateBytes = (size_t)B_ * C_ * T_ * T_ * sizeof(float);    // 8 MB
    const size_t zupBytes   = (size_t)B_ * L_ * T_ * T_ * sizeof(_Float16); // 8 MB
    float*    stateA = (float*)ws;
    float*    stateB = (float*)(ws + stateBytes);
    _Float16* zupH   = (_Float16*)(ws + 2 * stateBytes);
    _Float16* w1s    = (_Float16*)(ws + 2 * stateBytes + zupBytes);
    _Float16* w2s    = (_Float16*)(ws + 2 * stateBytes + zupBytes +
                                   (size_t)96 * HID_ * sizeof(_Float16));

    nca_upsample<<<(B_ * L_ * T_ * T_) / 256, 256, 0, stream>>>(z_field, zupH);
    nca_wconv<<<(96 * HID_ + HID_ * C_ + 255) / 256, 256, 0, stream>>>(w1, w2, w1s, w2s);
    nca_init<<<(B_ * C_ * T_ * T_ / 4) / 256, 256, 0, stream>>>(stateA);

    for (int s = 0; s < STEPS_; ++s) {
        nca_step<<<dim3(T_, B_), 256, 0, stream>>>(stateA, stateB, zupH, w1s, w2s, b1, b2, s);
        nca_alive<<<(B_ * T_ * T_ / 4) / 256, 256, 0, stream>>>(stateB, stateA);
    }

    nca_out<<<(B_ * 3 * T_ * T_ / 4) / 256, 256, 0, stream>>>(stateA, out);
}
